// EncoderBlock_13477607375275
// MI455X (gfx1250) — compile-verified, matrix operand order checked by reading, not run on hardware
//
#include <hip/hip_runtime.h>
#include <hip/hip_bf16.h>

typedef _Float16 f16;
typedef __attribute__((ext_vector_type(16))) _Float16 v16h;
typedef __attribute__((ext_vector_type(8)))  _Float16 v8h;
typedef __attribute__((ext_vector_type(8)))  float    v8f;
typedef __attribute__((ext_vector_type(4)))  unsigned int u32x4;
typedef __attribute__((ext_vector_type(8)))  int i32x8;
typedef __attribute__((ext_vector_type(4)))  int i32x4;

#define DM   1024
#define DFF  4096
#define SEQ  2048
#define NB   4
#define NH   16
#define DK   64

// ---------------------------------------------------------------------------
// WMMA helpers (CDNA5 wave32, V_WMMA_F32_16X16X32_F16)
// ---------------------------------------------------------------------------
__device__ __forceinline__ v8f wmma16(v16h a, v16h b, v8f c) {
  return __builtin_amdgcn_wmma_f32_16x16x32_f16(
      /*neg_a=*/false, a, /*neg_b=*/false, b,
      /*c_mod=*/(short)0, c, /*reuse_a=*/false, /*reuse_b=*/false);
}

// 16x32 f16 fragment (A layout; B identical with N-major tile). Lane L holds
// row (L&15); two contiguous 8-half (16B) chunks per lane -> two ds_load_b128.
__device__ __forceinline__ v16h load_frag(const f16* base, int ld,
                                          int rowBase, int kBase, int lane) {
  const int r = rowBase + (lane & 15);
  const int g = (lane >> 4) << 3;              // 0 or 8
  const f16* p = base + r * ld + kBase + g;
  v8h lo = *(const v8h*)(p);
  v8h hi = *(const v8h*)(p + 16);
  v16h out;
#pragma unroll
  for (int i = 0; i < 8; ++i) { out[i] = lo[i]; out[i + 8] = hi[i]; }
  return out;
}

// ---------------------------------------------------------------------------
// Tensor Data Mover: DMA a 2D f16 tile [tileRows x tileCols] from a row-major
// global tensor (row length = rowLen halves) into LDS, adding padAmt(+1)
// DWORDs of LDS padding after every 2<<padInt DWORDs (= one tile row).
// D# layout per CDNA5 ISA. Issued by one wave; tracked
// with TENSORcnt. Low 32 bits of a flat __shared__ pointer are the LDS byte
// address.
// This toolchain exposes the 6-arg builtin:
//   (u32x4 g0, i32x8 g1, i32x4 g2, i32x4 g3, i32x8 extra, i32 cpol)
// ---------------------------------------------------------------------------
__device__ __forceinline__ void tdm_load_2d(const f16* gsrc, const f16* ldst,
                                            int rowLen, int tileCols,
                                            int tileRows, int padInt, int padAmt) {
  const unsigned long long ga = (unsigned long long)(size_t)gsrc;
  const unsigned lds = (unsigned)(size_t)ldst;
  u32x4 g0;
  g0[0] = 1u;                                                  // count=1, user D#
  g0[1] = lds;                                                 // lds_addr (bytes)
  g0[2] = (unsigned)ga;                                        // global_addr[31:0]
  g0[3] = (unsigned)((ga >> 32) & 0x01ffffffu) | (2u << 30);   // ga[56:32] | type=2
  i32x8 g1;
  g1[0] = (int)((1u << 16)                      // data_size = 1 -> 2 bytes
              | (1u << 20)                      // pad_enable
              | ((unsigned)padInt << 22)        // pad_interval code
              | ((unsigned)padAmt << 25));      // pad_amount code
  g1[1] = (int)(((unsigned)rowLen & 0xffffu) << 16);                 // dim0[15:0]
  g1[2] = (int)((((unsigned)rowLen >> 16) & 0xffffu)
              | (((unsigned)tileRows & 0xffffu) << 16));             // dim0 hi | dim1 lo
  g1[3] = (int)(((unsigned)tileCols & 0xffffu) << 16);               // dim1 hi=0 | tile_dim0
  g1[4] = (int)((unsigned)tileRows & 0xffffu);                       // tile_dim1 | tile_dim2=0
  g1[5] = (int)rowLen;                                               // dim0_stride[31:0]
  g1[6] = 0;                                                         // stride hi | dim1_stride lo
  g1[7] = 0;                                                         // dim1_stride hi
  const i32x4 z4 = {0, 0, 0, 0};
  const i32x8 z8 = {0, 0, 0, 0, 0, 0, 0, 0};
  __builtin_amdgcn_tensor_load_to_lds(g0, g1, z4, z4, z8, 0);
}

// ---------------------------------------------------------------------------
// Weight convert + transpose: f32 [K][N] row-major -> f16 [N][K] row-major
// ---------------------------------------------------------------------------
__global__ void __launch_bounds__(256)
transpose_w(const float* __restrict__ in, f16* __restrict__ out, int K, int N) {
  __shared__ float tile[32][33];
  const int kb = blockIdx.y * 32;
  const int nb = blockIdx.x * 32;
  const int tx = threadIdx.x, ty = threadIdx.y;
#pragma unroll
  for (int i = 0; i < 32; i += 8)
    tile[ty + i][tx] = in[(size_t)(kb + ty + i) * N + nb + tx];
  __syncthreads();
#pragma unroll
  for (int i = 0; i < 32; i += 8)
    out[(size_t)(nb + ty + i) * K + kb + tx] = (f16)tile[tx][ty + i];
}

// ---------------------------------------------------------------------------
// LayerNorm (Bessel-corrected var, eps added to std) -> f16 activations
// ---------------------------------------------------------------------------
__global__ void __launch_bounds__(256)
layernorm_f16(const float* __restrict__ x, const float* __restrict__ gain,
              const float* __restrict__ bias, f16* __restrict__ out) {
  const int row = blockIdx.x;
  const float* xr = x + (size_t)row * DM;
  const int t = threadIdx.x;
  const float4 v = *(const float4*)(xr + t * 4);
  float s  = v.x + v.y + v.z + v.w;
  float ss = v.x * v.x + v.y * v.y + v.z * v.z + v.w * v.w;
#pragma unroll
  for (int off = 16; off; off >>= 1) {
    s  += __shfl_xor(s,  off, 32);
    ss += __shfl_xor(ss, off, 32);
  }
  __shared__ float rs[8], rss[8], fin[2];
  if ((t & 31) == 0) { rs[t >> 5] = s; rss[t >> 5] = ss; }
  __syncthreads();
  if (t == 0) {
    float a = 0.f, b = 0.f;
    for (int i = 0; i < 8; ++i) { a += rs[i]; b += rss[i]; }
    fin[0] = a; fin[1] = b;
  }
  __syncthreads();
  const float mean = fin[0] * (1.f / DM);
  float var = fin[1] * (1.f / DM) - mean * mean;
  var *= (float)DM / (float)(DM - 1);
  const float inv = 1.f / (sqrtf(var) + 1e-6f);
  const float4 g = *(const float4*)(gain + t * 4);
  const float4 b4 = *(const float4*)(bias + t * 4);
  f16* o = out + (size_t)row * DM + t * 4;
  o[0] = (f16)((v.x - mean) * inv * g.x + b4.x);
  o[1] = (f16)((v.y - mean) * inv * g.y + b4.y);
  o[2] = (f16)((v.z - mean) * inv * g.z + b4.z);
  o[3] = (f16)((v.w - mean) * inv * g.w + b4.w);
}

// ---------------------------------------------------------------------------
// WMMA GEMM with TDM-staged, double-buffered LDS tiles.
// C[M,N] = A[M,K](f16) * Bt[N,K](f16) + bias, fused epilogue.
// flags: 1 = ReLU, 2 = QKV head-scatter f16 output, 4 = scale by 1/sqrt(DK)
// 256 threads = 8 waves; block tile 128x128, K-step 32; wave tile 32x64.
// Wave 0 issues tensor_load_to_lds for tile t+1 while all waves compute t.
// ---------------------------------------------------------------------------
__global__ void __launch_bounds__(256)
gemm_wmma(const f16* __restrict__ A, const f16* __restrict__ Bt,
          const float* __restrict__ bias,
          f16* __restrict__ outH, float* __restrict__ outF,
          const float* __restrict__ resid,
          int M, int N, int K, int flags) {
  __shared__ __align__(16) f16 As[2][128 * 40];   // 32-half rows + 8-half pad
  __shared__ __align__(16) f16 Bs[2][128 * 40];

  const int m0 = blockIdx.y * 128;
  const int n0 = blockIdx.x * 128;
  const int lane = threadIdx.x & 31;
  const int wave = threadIdx.x >> 5;
  const int mW = (wave & 3) * 32;
  const int nW = (wave >> 2) * 64;
  const bool issuer = (threadIdx.x < 32);        // wave 0 drives the TDM

  const v8f zero8 = {0.f, 0.f, 0.f, 0.f, 0.f, 0.f, 0.f, 0.f};
  v8f acc[2][4];
#pragma unroll
  for (int i = 0; i < 2; ++i)
#pragma unroll
    for (int j = 0; j < 4; ++j) acc[i][j] = zero8;

  const int T = K / 32;
  // prologue: stage tile 0. Row = 32 halves = 64B = 16 DW -> padInt=3;
  // pad 16B = 4 DW -> padAmt=3  => LDS row stride 40 halves.
  if (issuer) {
    tdm_load_2d(A  + (size_t)m0 * K, &As[0][0], K, 32, 128, 3, 3);
    tdm_load_2d(Bt + (size_t)n0 * K, &Bs[0][0], K, 32, 128, 3, 3);
  }

  for (int t = 0; t < T; ++t) {
    __syncthreads();                             // everyone done with buf[(t-1)&1]
    if (issuer) {
      if (t + 1 < T) {
        const int k1 = (t + 1) * 32;
        tdm_load_2d(A  + (size_t)m0 * K + k1, &As[(t + 1) & 1][0], K, 32, 128, 3, 3);
        tdm_load_2d(Bt + (size_t)n0 * K + k1, &Bs[(t + 1) & 1][0], K, 32, 128, 3, 3);
        __builtin_amdgcn_s_wait_tensorcnt(2);    // tile t landed; t+1 in flight
      } else {
        __builtin_amdgcn_s_wait_tensorcnt(0);
      }
    }
    __syncthreads();                             // publish tile t to all waves

    const f16* Ab = &As[t & 1][0];
    const f16* Bb = &Bs[t & 1][0];
    v16h a0 = load_frag(Ab, 40, mW,      0, lane);
    v16h a1 = load_frag(Ab, 40, mW + 16, 0, lane);
#pragma unroll
    for (int j = 0; j < 4; ++j) {
      v16h b = load_frag(Bb, 40, nW + j * 16, 0, lane);
      acc[0][j] = wmma16(a0, b, acc[0][j]);
      acc[1][j] = wmma16(a1, b, acc[1][j]);
    }
  }

  const int col = lane & 15;
  const int rh  = (lane >> 4) << 3;
#pragma unroll
  for (int j = 0; j < 4; ++j) {
    const int n = n0 + nW + j * 16 + col;
    const float bv = bias ? bias[n] : 0.f;
#pragma unroll
    for (int i = 0; i < 2; ++i) {
#pragma unroll
      for (int v = 0; v < 8; ++v) {
        const int m = m0 + mW + i * 16 + v + rh;
        float val = acc[i][j][v] + bv;
        if (flags & 1) val = fmaxf(val, 0.f);
        if (flags & 4) val *= 0.125f;            // 1/sqrt(DK), folded into Q
        if (outF) {
          const float r = resid ? resid[(size_t)m * N + n] : 0.f;
          outF[(size_t)m * N + n] = val + r;
        }
        if (outH) {
          size_t o;
          if (flags & 2) {                       // [B,H,S,DK] scatter
            o = (((size_t)(m >> 11) * NH + (n >> 6)) * SEQ + (m & (SEQ - 1))) * DK
                + (n & (DK - 1));
          } else {
            o = (size_t)m * N + n;
          }
          outH[o] = (f16)val;
        }
      }
    }
  }
}

// ---------------------------------------------------------------------------
// Flash attention: grid (B*H, S/128); 256 threads = 8 waves; each wave owns
// 16 query rows. Q/K tiles staged by TDM; V loaded+transposed manually
// (TDM cannot transpose). Online softmax; P staged through LDS (same-wave
// LDS ordering makes the P write->read barrier-free).
// ---------------------------------------------------------------------------
__global__ void __launch_bounds__(256)
attn_flash(const f16* __restrict__ Q, const f16* __restrict__ Kc,
           const f16* __restrict__ V, const int* __restrict__ mask,
           f16* __restrict__ ctx) {
  __shared__ __align__(16) f16 Qs[128 * 72];
  __shared__ __align__(16) f16 Ks[128 * 72];
  __shared__ __align__(16) f16 Vt[64 * 136];    // V transposed: [d][key]
  __shared__ __align__(16) f16 Ps[128 * 136];

  const int bh = blockIdx.x;
  const int b  = bh >> 4;                       // H = 16
  const int h  = bh & 15;
  const int q0 = blockIdx.y * 128;
  const int lane = threadIdx.x & 31;
  const int wave = threadIdx.x >> 5;
  const int col  = lane & 15;
  const int rh   = (lane >> 4) << 3;
  const bool issuer = (threadIdx.x < 32);

  const f16* Qh = Q  + ((size_t)bh * SEQ + q0) * DK;
  const f16* Kh = Kc + (size_t)bh * SEQ * DK;
  const f16* Vh = V  + (size_t)bh * SEQ * DK;

  // Q tile 128x64: row = 64 halves = 128B = 32 DW -> padInt=4; pad 4 DW -> padAmt=3
  // => LDS row stride 72 halves.
  if (issuer) {
    tdm_load_2d(Qh, Qs, DK, 64, 128, 4, 3);
    __builtin_amdgcn_s_wait_tensorcnt(0);
  }

  float mrow[8], lrow[8], sc[8];
#pragma unroll
  for (int v = 0; v < 8; ++v) { mrow[v] = -1e30f; lrow[v] = 0.f; }
  const v8f zero8 = {0.f, 0.f, 0.f, 0.f, 0.f, 0.f, 0.f, 0.f};
  v8f O[4];
#pragma unroll
  for (int j = 0; j < 4; ++j) O[j] = zero8;

  __syncthreads();
  const v16h qa0 = load_frag(Qs, 72, wave * 16, 0,  lane);
  const v16h qa1 = load_frag(Qs, 72, wave * 16, 32, lane);

  for (int kv = 0; kv < SEQ; kv += 128) {
    __syncthreads();                             // prior reads of Ks/Vt done
    if (issuer)
      tdm_load_2d(Kh + (size_t)kv * DK, Ks, DK, 64, 128, 4, 3);
    {
      const int row = threadIdx.x >> 1;
      const int cB  = (threadIdx.x & 1) * 32;
#pragma unroll
      for (int c = 0; c < 32; c += 8) {
        v8h tv = *(const v8h*)(Vh + (size_t)(kv + row) * DK + cB + c);
#pragma unroll
        for (int e = 0; e < 8; ++e) Vt[(cB + c + e) * 136 + row] = tv[e];
      }
    }
    if (issuer) __builtin_amdgcn_s_wait_tensorcnt(0);
    __syncthreads();

    // S = Q * K^T  (Q pre-scaled by 1/sqrt(DK) at projection time)
    v8f s[8];
#pragma unroll
    for (int j = 0; j < 8; ++j) s[j] = zero8;
#pragma unroll
    for (int j = 0; j < 8; ++j) {
      v16h b0 = load_frag(Ks, 72, j * 16, 0,  lane);
      v16h b1 = load_frag(Ks, 72, j * 16, 32, lane);
      s[j] = wmma16(qa0, b0, s[j]);
      s[j] = wmma16(qa1, b1, s[j]);
    }

    // mask: scores where mask==0 -> -1e9
#pragma unroll
    for (int j = 0; j < 8; ++j) {
      if (mask[b * SEQ + kv + j * 16 + col] == 0) {
#pragma unroll
        for (int v = 0; v < 8; ++v) s[j][v] = -1e9f;
      }
    }

    // online softmax; each 16-lane half holds one row per C-layout VGPR
#pragma unroll
    for (int v = 0; v < 8; ++v) {
      float tmax = s[0][v];
#pragma unroll
      for (int j = 1; j < 8; ++j) tmax = fmaxf(tmax, s[j][v]);
#pragma unroll
      for (int off = 1; off < 16; off <<= 1)
        tmax = fmaxf(tmax, __shfl_xor(tmax, off, 32));
      const float mnew = fmaxf(mrow[v], tmax);
      sc[v] = __expf(mrow[v] - mnew);
      float rsum = 0.f;
#pragma unroll
      for (int j = 0; j < 8; ++j) {
        const float p = __expf(s[j][v] - mnew);
        s[j][v] = p;
        rsum += p;
      }
#pragma unroll
      for (int off = 1; off < 16; off <<= 1) rsum += __shfl_xor(rsum, off, 32);
      lrow[v] = lrow[v] * sc[v] + rsum;
      mrow[v] = mnew;
    }

    // C-layout -> A-layout through LDS (same-wave rows: LDS stays in order)
#pragma unroll
    for (int v = 0; v < 8; ++v) {
      const int r = wave * 16 + v + rh;
#pragma unroll
      for (int j = 0; j < 8; ++j) Ps[r * 136 + j * 16 + col] = (f16)s[j][v];
    }

#pragma unroll
    for (int jd = 0; jd < 4; ++jd)
#pragma unroll
      for (int v = 0; v < 8; ++v) O[jd][v] *= sc[v];

    // O += P * V
#pragma unroll
    for (int kk = 0; kk < 128; kk += 32) {
      v16h pa = load_frag(Ps, 136, wave * 16, kk, lane);
#pragma unroll
      for (int jd = 0; jd < 4; ++jd) {
        v16h bv = load_frag(Vt, 136, jd * 16, kk, lane);
        O[jd] = wmma16(pa, bv, O[jd]);
      }
    }
  }

  // normalize and write context in [B*S, H*DK] layout for the out-projection
#pragma unroll
  for (int v = 0; v < 8; ++v) {
    const float inv = 1.f / lrow[v];
    const int r = q0 + wave * 16 + v + rh;
    const size_t m = (size_t)b * SEQ + r;
#pragma unroll
    for (int jd = 0; jd < 4; ++jd)
      ctx[m * DM + h * DK + jd * 16 + col] = (f16)(O[jd][v] * inv);
  }
}

// ---------------------------------------------------------------------------
// Launch
// ---------------------------------------------------------------------------
extern "C" void kernel_launch(void* const* d_in, const int* in_sizes, int n_in,
                              void* d_out, int out_size, void* d_ws, size_t ws_size,
                              hipStream_t stream) {
  (void)in_sizes; (void)n_in; (void)out_size; (void)ws_size;

  const float* x    = (const float*)d_in[0];
  const int*   mask = (const int*)  d_in[1];
  const float* Wq = (const float*)d_in[2];  const float* bq = (const float*)d_in[3];
  const float* Wk = (const float*)d_in[4];  const float* bk = (const float*)d_in[5];
  const float* Wv = (const float*)d_in[6];  const float* bv = (const float*)d_in[7];
  const float* Wo = (const float*)d_in[8];  const float* bo = (const float*)d_in[9];
  const float* W1 = (const float*)d_in[10]; const float* b1 = (const float*)d_in[11];
  const float* W2 = (const float*)d_in[12]; const float* b2 = (const float*)d_in[13];
  const float* g1 = (const float*)d_in[14]; const float* bi1 = (const float*)d_in[15];
  const float* g2 = (const float*)d_in[16]; const float* bi2 = (const float*)d_in[17];
  float* out = (float*)d_out;

  const size_t M = (size_t)NB * SEQ;  // 8192 rows

  char* base = (char*)d_ws;
  size_t off = 0;
  auto alloc = [&](size_t bytes) -> void* {
    void* p = base + off;
    off += (bytes + 255) & ~(size_t)255;
    return p;
  };
  f16*   xn  = (f16*)  alloc(M * DM * 2);        // ln1 out, reused for ln2 out
  f16*   WqT = (f16*)  alloc((size_t)DM * DM * 2);
  f16*   WkT = (f16*)  alloc((size_t)DM * DM * 2);
  f16*   WvT = (f16*)  alloc((size_t)DM * DM * 2);
  f16*   WoT = (f16*)  alloc((size_t)DM * DM * 2);
  f16*   W1T = (f16*)  alloc((size_t)DM * DFF * 2);
  f16*   W2T = (f16*)  alloc((size_t)DM * DFF * 2);
  f16*   q16 = (f16*)  alloc(M * DM * 2);        // [B,H,S,DK]
  f16*   k16 = (f16*)  alloc(M * DM * 2);
  f16*   v16 = (f16*)  alloc(M * DM * 2);
  f16*   ctx = (f16*)  alloc(M * DM * 2);
  float* y1  = (float*)alloc(M * DM * 4);        // x + attn (fp32 residual)
  f16*   h16 = (f16*)  alloc(M * DFF * 2);

  dim3 t32(32, 8);
  transpose_w<<<dim3(DM / 32, DM / 32),  t32, 0, stream>>>(Wq, WqT, DM, DM);
  transpose_w<<<dim3(DM / 32, DM / 32),  t32, 0, stream>>>(Wk, WkT, DM, DM);
  transpose_w<<<dim3(DM / 32, DM / 32),  t32, 0, stream>>>(Wv, WvT, DM, DM);
  transpose_w<<<dim3(DM / 32, DM / 32),  t32, 0, stream>>>(Wo, WoT, DM, DM);
  transpose_w<<<dim3(DFF / 32, DM / 32), t32, 0, stream>>>(W1, W1T, DM, DFF);
  transpose_w<<<dim3(DM / 32, DFF / 32), t32, 0, stream>>>(W2, W2T, DFF, DM);

  layernorm_f16<<<(unsigned)M, 256, 0, stream>>>(x, g1, bi1, xn);

  const dim3 gDM(DM / 128, (unsigned)(M / 128));
  gemm_wmma<<<gDM, 256, 0, stream>>>(xn, WqT, bq, q16, nullptr, nullptr,
                                     (int)M, DM, DM, 2 | 4);
  gemm_wmma<<<gDM, 256, 0, stream>>>(xn, WkT, bk, k16, nullptr, nullptr,
                                     (int)M, DM, DM, 2);
  gemm_wmma<<<gDM, 256, 0, stream>>>(xn, WvT, bv, v16, nullptr, nullptr,
                                     (int)M, DM, DM, 2);

  attn_flash<<<dim3(NB * NH, SEQ / 128), 256, 0, stream>>>(q16, k16, v16, mask, ctx);

  gemm_wmma<<<gDM, 256, 0, stream>>>(ctx, WoT, bo, nullptr, y1, x,
                                     (int)M, DM, DM, 0);

  layernorm_f16<<<(unsigned)M, 256, 0, stream>>>(y1, g2, bi2, xn);

  gemm_wmma<<<dim3(DFF / 128, (unsigned)(M / 128)), 256, 0, stream>>>(
      xn, W1T, b1, h16, nullptr, nullptr, (int)M, DFF, DM, 1);
  gemm_wmma<<<gDM, 256, 0, stream>>>(h16, W2T, b2, nullptr, out, y1,
                                     (int)M, DM, DFF, 0);
}
